// GATEncoder_15006615734310
// MI455X (gfx1250) — matrix operand order repaired and checked by reading, not compiled, not run on hardware
//
#include <hip/hip_runtime.h>
#include <hip/hip_bf16.h>

// ---------------------------------------------------------------------------
// Types for CDNA5 WMMA (wave32): v_wmma_f32_16x16x32_bf16
// ---------------------------------------------------------------------------
typedef __bf16 bf16_t;
typedef __attribute__((ext_vector_type(16))) __bf16 v16bf;
typedef __attribute__((ext_vector_type(8)))  __bf16 v8bf;
typedef __attribute__((ext_vector_type(8)))  float  v8f;

#define N_NODES 50000
#define N_EDGES 600000
#define HEADS   3
#define SLOPE   0.15f

// ---------------------------------------------------------------------------
// Small utility kernels
// ---------------------------------------------------------------------------
__global__ void k_fill_u32(unsigned* __restrict__ p, unsigned val, int n) {
  int i = blockIdx.x * blockDim.x + threadIdx.x;
  if (i < n) p[i] = val;
}

__global__ void k_f32_to_bf16(const float* __restrict__ s, bf16_t* __restrict__ d, int n) {
  int i = blockIdx.x * blockDim.x + threadIdx.x;
  if (i < n) d[i] = (bf16_t)s[i];
}

// Wt[m*K + k] = W[k*M_full + col0 + m]   (bf16 transpose of an f32 [K, M_full] slice)
__global__ void k_transpose_bf16(const float* __restrict__ W, bf16_t* __restrict__ Wt,
                                 int K, int M_full, int col0, int M) {
  int i = blockIdx.x * blockDim.x + threadIdx.x;
  if (i >= M * K) return;
  int m = i / K, k = i % K;
  Wt[(long)m * K + k] = (bf16_t)W[(long)k * M_full + col0 + m];
}

// Block-diagonal builder: w is [E, bk, bm] f32; output Wt is [M=E*bm, K=E*bk] bf16
// representing the transposed block-diagonal matrix BD[k, m].
__global__ void k_blockdiag_bf16(const float* __restrict__ w, bf16_t* __restrict__ Wt,
                                 int E, int bk, int bm) {
  int K = E * bk, M = E * bm;
  int i = blockIdx.x * blockDim.x + threadIdx.x;
  if (i >= M * K) return;
  int m = i / K, k = i % K;
  int e = m / bm;
  float v = 0.f;
  if (k / bk == e) v = w[((long)e * bk + (k - e * bk)) * bm + (m - e * bm)];
  Wt[(long)m * K + k] = (bf16_t)v;
}

// rexp_w1 [8,64,64] (e,k,h) -> dense Wt [M=512, K=64]: Wt[(e*64+h)*64 + k]
__global__ void k_prep_rbd1(const float* __restrict__ w, bf16_t* __restrict__ Wt) {
  int i = blockIdx.x * blockDim.x + threadIdx.x;  // over 512*64
  if (i >= 512 * 64) return;
  int m = i / 64, k = i % 64;
  int e = m / 64, h = m % 64;
  Wt[(long)m * 64 + k] = (bf16_t)w[(long)e * 4096 + (long)k * 64 + h];
}

// ---------------------------------------------------------------------------
// WMMA GEMM: C[N,M] = act(A[N,K](bf16) * W[K,M] + bias), W given transposed
// as Wt[M,K] bf16. One wave computes a 16x64 strip (4 accumulators).
// Requires: N % 16 == 0, K % 32 == 0, M % 64 == 0.
// ---------------------------------------------------------------------------
#define GEMM_WPB 8   // waves per block (block = 256 threads)

__global__ void k_wmma_gemm(const bf16_t* __restrict__ A, const bf16_t* __restrict__ Wt,
                            const float* __restrict__ bias,
                            float* __restrict__ Cf, bf16_t* __restrict__ Cb,
                            int N, int K, int M, int relu) {
  const int lane = threadIdx.x & 31;
  const int wave = threadIdx.x >> 5;
  const int rowTile = blockIdx.x * GEMM_WPB + wave;
  if (rowTile * 16 >= N) return;
  const int col0 = blockIdx.y * 64;
  const int mlane = lane & 15;   // A: row within tile, B: col within tile
  const int g     = lane >> 4;   // lane half selects K sub-range
  const long arow = (long)(rowTile * 16 + mlane) * K;

  v8f acc0 = {}, acc1 = {}, acc2 = {}, acc3 = {};

  for (int kb = 0; kb < K; kb += 32) {
    // A fragment: 16-bit A 16x32 layout (ISA 7.12.2): lanes 0-15 K=0..7,16..23;
    // lanes 16-31 K=8..15,24..31 -> two contiguous 16B loads per lane.
    v8bf alo = *(const v8bf*)(A + arow + kb + g * 8);
    v8bf ahi = *(const v8bf*)(A + arow + kb + 16 + g * 8);
    v16bf a;
#pragma unroll
    for (int i = 0; i < 8; i++) { a[i] = alo[i]; a[i + 8] = ahi[i]; }

#pragma unroll
    for (int t = 0; t < 4; t++) {
      // B fragment: 32x16 layout matching A: lanes 0-15 K=0..7,16..23,
      // lanes 16-31 K=8..15,24..31 -> two 16B loads from Wt row (W column).
      const bf16_t* wp = Wt + (long)(col0 + t * 16 + mlane) * K + kb + g * 8;
      v8bf blo = *(const v8bf*)(wp);
      v8bf bhi = *(const v8bf*)(wp + 16);
      v16bf b;
#pragma unroll
      for (int i = 0; i < 8; i++) { b[i] = blo[i]; b[i + 8] = bhi[i]; }

      v8f* accp = (t == 0) ? &acc0 : (t == 1) ? &acc1 : (t == 2) ? &acc2 : &acc3;
      *accp = __builtin_amdgcn_wmma_f32_16x16x32_bf16(
          /*neg_a=*/false, a, /*neg_b=*/false, b,
          /*c_mod=*/(short)0, *accp, /*reuse_a=*/false, /*reuse_b=*/false);
    }
  }

  // C layout: lane n = lane&15, VGPR v holds row v + 8*(lane>>4)
  const int ncol = lane & 15;
  const int rhalf = lane >> 4;
#pragma unroll
  for (int t = 0; t < 4; t++) {
    const v8f acc = (t == 0) ? acc0 : (t == 1) ? acc1 : (t == 2) ? acc2 : acc3;
    const int col = col0 + t * 16 + ncol;
    const float bv = bias ? bias[col] : 0.f;
#pragma unroll
    for (int v = 0; v < 8; v++) {
      const int row = rowTile * 16 + rhalf * 8 + v;
      float val = acc[v] + bv;
      if (relu && val < 0.f) val = 0.f;
      if (Cf) Cf[(long)row * M + col] = val;
      if (Cb) Cb[(long)row * M + col] = (bf16_t)val;
    }
  }
}

// ---------------------------------------------------------------------------
// Edge-wise GATv2 passes (segment softmax via atomics)
// ---------------------------------------------------------------------------
__device__ __forceinline__ unsigned fenc(float f) {
  unsigned u = __float_as_uint(f);
  return (u & 0x80000000u) ? ~u : (u | 0x80000000u);
}
__device__ __forceinline__ float fdec(unsigned u) {
  return __uint_as_float((u & 0x80000000u) ? (u & 0x7fffffffu) : ~u);
}

__global__ void k_edge_score(const float* __restrict__ xl, const float* __restrict__ xr,
                             const float* __restrict__ att,
                             const int* __restrict__ src, const int* __restrict__ dst,
                             float* __restrict__ s, unsigned* __restrict__ menc,
                             int E, int H, int C) {
  int i = blockIdx.x * blockDim.x + threadIdx.x;
  if (i >= E * H) return;
  int e = i / H, h = i % H;
  const float* pl = xl + (long)src[e] * H * C + (long)h * C;
  const float* pr = xr + (long)dst[e] * H * C + (long)h * C;
  const float* pa = att + (long)h * C;
  float acc = 0.f;
  for (int c = 0; c < C; c++) {
    float v = pl[c] + pr[c];
    v = (v >= 0.f) ? v : SLOPE * v;
    acc += pa[c] * v;
  }
  s[i] = acc;
  atomicMax(&menc[(long)dst[e] * H + h], fenc(acc));
}

__global__ void k_decode_max(unsigned* __restrict__ menc, int n) {
  int i = blockIdx.x * blockDim.x + threadIdx.x;
  if (i >= n) return;
  float v = fdec(menc[i]);
  if (!isfinite(v)) v = 0.f;
  ((float*)menc)[i] = v;
}

__global__ void k_edge_exp(float* __restrict__ s, const float* __restrict__ m,
                           const int* __restrict__ dst, float* __restrict__ den,
                           int E, int H) {
  int i = blockIdx.x * blockDim.x + threadIdx.x;
  if (i >= E * H) return;
  int e = i / H, h = i % H;
  float a = __expf(s[i] - m[(long)dst[e] * H + h]);
  s[i] = a;
  atomicAdd(&den[(long)dst[e] * H + h], a);
}

__global__ void k_edge_agg(const float* __restrict__ a, const float* __restrict__ den,
                           const float* __restrict__ xl,
                           const int* __restrict__ src, const int* __restrict__ dst,
                           float* __restrict__ agg, int E, int H, int C) {
  int i = blockIdx.x * blockDim.x + threadIdx.x;
  if (i >= E * H) return;
  int e = i / H, h = i % H;
  float w = a[i] / (den[(long)dst[e] * H + h] + 1e-16f);
  float* pagg = agg + (long)dst[e] * H * C + (long)h * C;
  const float* pl = xl + (long)src[e] * H * C + (long)h * C;
  for (int c = 0; c < C; c++) atomicAdd(&pagg[c], w * pl[c]);
}

// ---------------------------------------------------------------------------
// Elementwise glue kernels
// ---------------------------------------------------------------------------
__global__ void k_build_hcat(const float* __restrict__ x, const float* __restrict__ agg1,
                             const float* __restrict__ bias1, bf16_t* __restrict__ h, int N) {
  int i = blockIdx.x * blockDim.x + threadIdx.x;
  if (i >= N * 256) return;
  int n = i / 256, j = i % 256;
  float v;
  if (j < 64) v = x[(long)n * 64 + j];
  else {
    int c = j - 64;
    v = agg1[(long)n * 192 + c] + bias1[c];
    if (v < 0.f) v = 0.f;
  }
  h[i] = (bf16_t)v;
}

__global__ void k_head_mean_relu(const float* __restrict__ agg2, const float* __restrict__ bias2,
                                 float* __restrict__ h2, bf16_t* __restrict__ h2b, int N) {
  int i = blockIdx.x * blockDim.x + threadIdx.x;
  if (i >= N * 128) return;
  int n = i / 128, c = i % 128;
  float v = (agg2[(long)n * 384 + c] + agg2[(long)n * 384 + 128 + c] +
             agg2[(long)n * 384 + 256 + c]) * (1.f / 3.f) + bias2[c];
  if (v < 0.f) v = 0.f;
  h2[i] = v;
  h2b[i] = (bf16_t)v;
}

__global__ void k_layernorm64(const float* __restrict__ in, const float* __restrict__ g,
                              const float* __restrict__ b, float* __restrict__ out,
                              bf16_t* __restrict__ outb, int N) {
  int row = blockIdx.x * (blockDim.x >> 5) + (threadIdx.x >> 5);
  if (row >= N) return;
  int lane = threadIdx.x & 31;
  float v0 = in[(long)row * 64 + lane];
  float v1 = in[(long)row * 64 + 32 + lane];
  float s = v0 + v1;
  for (int off = 16; off; off >>= 1) s += __shfl_xor(s, off);
  float mu = s * (1.f / 64.f);
  float d0 = v0 - mu, d1 = v1 - mu;
  float q = d0 * d0 + d1 * d1;
  for (int off = 16; off; off >>= 1) q += __shfl_xor(q, off);
  float rs = rsqrtf(q * (1.f / 64.f) + 1e-5f);
  float o0 = d0 * rs * g[lane] + b[lane];
  float o1 = d1 * rs * g[32 + lane] + b[32 + lane];
  out[(long)row * 64 + lane] = o0;
  out[(long)row * 64 + 32 + lane] = o1;
  outb[(long)row * 64 + lane] = (bf16_t)o0;
  outb[(long)row * 64 + 32 + lane] = (bf16_t)o1;
}

__global__ void k_sel_softmax(const float* __restrict__ attn, const float* __restrict__ sel_w,
                              const float* __restrict__ sel_b, float* __restrict__ ew, int N) {
  int n = blockIdx.x * blockDim.x + threadIdx.x;
  if (n >= N) return;
  float logit[8];
#pragma unroll
  for (int j = 0; j < 8; j++) logit[j] = sel_b[j];
  const float* pa = attn + (long)n * 64;
  for (int c = 0; c < 64; c++) {
    float a = pa[c];
#pragma unroll
    for (int j = 0; j < 8; j++) logit[j] += a * sel_w[c * 8 + j];
  }
  float mx = logit[0];
#pragma unroll
  for (int j = 1; j < 8; j++) mx = fmaxf(mx, logit[j]);
  float sum = 0.f;
#pragma unroll
  for (int j = 0; j < 8; j++) { logit[j] = __expf(logit[j] - mx); sum += logit[j]; }
  float inv = 1.f / sum;
#pragma unroll
  for (int j = 0; j < 8; j++) ew[(long)n * 8 + j] = logit[j] * inv;
}

__global__ void k_routed_combine(const float* __restrict__ eo, const float* __restrict__ ew,
                                 const float* __restrict__ x, float* __restrict__ out, int N) {
  int i = blockIdx.x * blockDim.x + threadIdx.x;
  if (i >= N * 128) return;
  int n = i / 128, j = i % 128;
  if (j < 64) {
    out[i] = x[(long)n * 64 + j];
  } else {
    int q = j - 64;
    const float* pe = eo + (long)n * 512;
    const float* pw = ew + (long)n * 8;
    float acc = 0.f;
#pragma unroll
    for (int e = 0; e < 8; e++) acc += pw[e] * pe[e * 64 + q];
    out[i] = acc;
  }
}

// ---------------------------------------------------------------------------
// Host-side launch
// ---------------------------------------------------------------------------
static inline int cdiv(int a, int b) { return (a + b - 1) / b; }

extern "C" void kernel_launch(void* const* d_in, const int* in_sizes, int n_in,
                              void* d_out, int out_size, void* d_ws, size_t ws_size,
                              hipStream_t stream) {
  const int N = N_NODES, E = N_EDGES;
  const float* x       = (const float*)d_in[0];
  const int*   eidx    = (const int*)d_in[1];
  const int*   src     = eidx;
  const int*   dst     = eidx + E;
  const float* W_l1    = (const float*)d_in[2];
  const float* b_l1    = (const float*)d_in[3];
  const float* W_r1    = (const float*)d_in[4];
  const float* b_r1    = (const float*)d_in[5];
  const float* att1    = (const float*)d_in[6];
  const float* bias1   = (const float*)d_in[7];
  const float* W_l2    = (const float*)d_in[8];
  const float* b_l2    = (const float*)d_in[9];
  const float* W_r2    = (const float*)d_in[10];
  const float* b_r2    = (const float*)d_in[11];
  const float* att2    = (const float*)d_in[12];
  const float* bias2   = (const float*)d_in[13];
  const float* exp_w1  = (const float*)d_in[14];
  const float* exp_b1  = (const float*)d_in[15];   // [8,64] flat = 512
  const float* exp_w2  = (const float*)d_in[16];
  const float* exp_b2  = (const float*)d_in[17];   // [8,16] flat = 128
  const float* v_w     = (const float*)d_in[22];
  const float* v_b     = (const float*)d_in[23];
  const float* in_w    = (const float*)d_in[24];   // [64,192]
  const float* in_b    = (const float*)d_in[25];   // [192]
  const float* out_w   = (const float*)d_in[26];
  const float* out_b   = (const float*)d_in[27];
  const float* ln_g    = (const float*)d_in[28];
  const float* ln_b    = (const float*)d_in[29];
  const float* sel_w   = (const float*)d_in[30];
  const float* sel_b   = (const float*)d_in[31];
  const float* rexp_w1 = (const float*)d_in[32];
  const float* rexp_b1 = (const float*)d_in[33];   // [8,64] flat = 512
  const float* rexp_w2 = (const float*)d_in[34];
  const float* rexp_b2 = (const float*)d_in[35];   // [8,64] flat = 512

  // ---- workspace bump allocator --------------------------------------------
  char* ws = (char*)d_ws;
  size_t off = 0;
  auto alloc = [&](size_t bytes) -> char* {
    char* p = ws + off;
    off = (off + bytes + 255) & ~(size_t)255;
    return p;
  };
  bf16_t* wt_l1  = (bf16_t*)alloc((size_t)192 * 64 * 2);
  bf16_t* wt_r1  = (bf16_t*)alloc((size_t)192 * 64 * 2);
  bf16_t* wt_l2  = (bf16_t*)alloc((size_t)384 * 256 * 2);
  bf16_t* wt_r2  = (bf16_t*)alloc((size_t)384 * 256 * 2);
  bf16_t* bd1t   = (bf16_t*)alloc((size_t)512 * 128 * 2);
  bf16_t* bd2t   = (bf16_t*)alloc((size_t)128 * 512 * 2);
  bf16_t* vwt    = (bf16_t*)alloc((size_t)64 * 128 * 2);
  bf16_t* inwvt  = (bf16_t*)alloc((size_t)64 * 64 * 2);
  bf16_t* outwt  = (bf16_t*)alloc((size_t)64 * 64 * 2);
  bf16_t* rbd1t  = (bf16_t*)alloc((size_t)512 * 64 * 2);
  bf16_t* rbd2t  = (bf16_t*)alloc((size_t)512 * 512 * 2);
  bf16_t* x_bf   = (bf16_t*)alloc((size_t)N * 64 * 2);
  bf16_t* h_cat  = (bf16_t*)alloc((size_t)N * 256 * 2);
  // Arena A: phase1 = xl1|xr1|agg1 (3 * N*192*4); phase2 = xl2|xr2 (2 * N*384*4)
  char* arenaA   = alloc((size_t)N * 768 * 4);
  float* xl1  = (float*)arenaA;
  float* xr1  = (float*)(arenaA + (size_t)N * 192 * 4);
  float* agg1 = (float*)(arenaA + (size_t)N * 384 * 4);
  float* xl2  = (float*)arenaA;                          // aliases xl1|xr1 (dead)
  float* xr2  = (float*)(arenaA + (size_t)N * 384 * 4);  // aliases agg1 (dead)
  // Arena B: agg2 (N*384*4) then eo (N*512*4)
  char* arenaB   = alloc((size_t)N * 512 * 4);
  float* agg2 = (float*)arenaB;
  float* eo   = (float*)arenaB;
  float*    sbuf   = (float*)alloc((size_t)E * HEADS * 4);     // edge scores -> a
  unsigned* menc   = (unsigned*)alloc((size_t)N * HEADS * 4);  // enc max -> f32 max
  float*    den    = (float*)alloc((size_t)N * HEADS * 4);
  float*    h2     = (float*)alloc((size_t)N * 128 * 4);
  bf16_t*   h2b    = (bf16_t*)alloc((size_t)N * 128 * 2);
  bf16_t*   s1e1   = (bf16_t*)alloc((size_t)N * 512 * 2);      // s1_bf then e1_bf
  bf16_t*   specb  = (bf16_t*)alloc((size_t)N * 128 * 2);
  bf16_t*   veb    = (bf16_t*)alloc((size_t)N * 64 * 2);
  bf16_t*   vpb    = (bf16_t*)alloc((size_t)N * 64 * 2);
  float*    mha    = (float*)alloc((size_t)N * 64 * 4);
  float*    attn   = (float*)alloc((size_t)N * 64 * 4);
  bf16_t*   attnb  = (bf16_t*)alloc((size_t)N * 64 * 2);
  float*    ew     = (float*)alloc((size_t)N * 8 * 4);
  (void)ws_size; (void)n_in; (void)in_sizes; (void)out_size;

  const int TB = 256;
  // ---- weight prep (tiny) --------------------------------------------------
  k_transpose_bf16<<<cdiv(192 * 64, TB), TB, 0, stream>>>(W_l1, wt_l1, 64, 192, 0, 192);
  k_transpose_bf16<<<cdiv(192 * 64, TB), TB, 0, stream>>>(W_r1, wt_r1, 64, 192, 0, 192);
  k_transpose_bf16<<<cdiv(384 * 256, TB), TB, 0, stream>>>(W_l2, wt_l2, 256, 384, 0, 384);
  k_transpose_bf16<<<cdiv(384 * 256, TB), TB, 0, stream>>>(W_r2, wt_r2, 256, 384, 0, 384);
  k_blockdiag_bf16<<<cdiv(512 * 128, TB), TB, 0, stream>>>(exp_w1, bd1t, 8, 16, 64);
  k_blockdiag_bf16<<<cdiv(128 * 512, TB), TB, 0, stream>>>(exp_w2, bd2t, 8, 64, 16);
  k_transpose_bf16<<<cdiv(64 * 128, TB), TB, 0, stream>>>(v_w, vwt, 128, 64, 0, 64);
  k_transpose_bf16<<<cdiv(64 * 64, TB), TB, 0, stream>>>(in_w, inwvt, 64, 192, 128, 64);
  k_transpose_bf16<<<cdiv(64 * 64, TB), TB, 0, stream>>>(out_w, outwt, 64, 64, 0, 64);
  k_prep_rbd1<<<cdiv(512 * 64, TB), TB, 0, stream>>>(rexp_w1, rbd1t);
  k_blockdiag_bf16<<<cdiv(512 * 512, TB), TB, 0, stream>>>(rexp_w2, rbd2t, 8, 64, 64);

  k_f32_to_bf16<<<cdiv(N * 64, TB), TB, 0, stream>>>(x, x_bf, N * 64);

  const int rowTiles = N / 16;                 // 3125
  const int gx = cdiv(rowTiles, GEMM_WPB);
  const unsigned NEG_INF_ENC = 0x007FFFFFu;    // fenc(-inf)

  // ---- GATv2 layer 1 -------------------------------------------------------
  k_wmma_gemm<<<dim3(gx, 192 / 64), TB, 0, stream>>>(x_bf, wt_l1, b_l1, xl1, nullptr, N, 64, 192, 0);
  k_wmma_gemm<<<dim3(gx, 192 / 64), TB, 0, stream>>>(x_bf, wt_r1, b_r1, xr1, nullptr, N, 64, 192, 0);

  k_fill_u32<<<cdiv(N * HEADS, TB), TB, 0, stream>>>(menc, NEG_INF_ENC, N * HEADS);
  hipMemsetAsync(den, 0, (size_t)N * HEADS * 4, stream);
  hipMemsetAsync(agg1, 0, (size_t)N * 192 * 4, stream);

  k_edge_score<<<cdiv(E * HEADS, TB), TB, 0, stream>>>(xl1, xr1, att1, src, dst, sbuf, menc, E, HEADS, 64);
  k_decode_max<<<cdiv(N * HEADS, TB), TB, 0, stream>>>(menc, N * HEADS);
  k_edge_exp<<<cdiv(E * HEADS, TB), TB, 0, stream>>>(sbuf, (const float*)menc, dst, den, E, HEADS);
  k_edge_agg<<<cdiv(E * HEADS, TB), TB, 0, stream>>>(sbuf, den, xl1, src, dst, agg1, E, HEADS, 64);

  k_build_hcat<<<cdiv(N * 256, TB), TB, 0, stream>>>(x, agg1, bias1, h_cat, N);

  // ---- GATv2 layer 2 -------------------------------------------------------
  k_wmma_gemm<<<dim3(gx, 384 / 64), TB, 0, stream>>>(h_cat, wt_l2, b_l2, xl2, nullptr, N, 256, 384, 0);
  k_wmma_gemm<<<dim3(gx, 384 / 64), TB, 0, stream>>>(h_cat, wt_r2, b_r2, xr2, nullptr, N, 256, 384, 0);

  k_fill_u32<<<cdiv(N * HEADS, TB), TB, 0, stream>>>(menc, NEG_INF_ENC, N * HEADS);
  hipMemsetAsync(den, 0, (size_t)N * HEADS * 4, stream);
  hipMemsetAsync(agg2, 0, (size_t)N * 384 * 4, stream);

  k_edge_score<<<cdiv(E * HEADS, TB), TB, 0, stream>>>(xl2, xr2, att2, src, dst, sbuf, menc, E, HEADS, 128);
  k_decode_max<<<cdiv(N * HEADS, TB), TB, 0, stream>>>(menc, N * HEADS);
  k_edge_exp<<<cdiv(E * HEADS, TB), TB, 0, stream>>>(sbuf, (const float*)menc, dst, den, E, HEADS);
  k_edge_agg<<<cdiv(E * HEADS, TB), TB, 0, stream>>>(sbuf, den, xl2, src, dst, agg2, E, HEADS, 128);

  k_head_mean_relu<<<cdiv(N * 128, TB), TB, 0, stream>>>(agg2, bias2, h2, h2b, N);

  // ---- partition experts as block-diagonal GEMMs ---------------------------
  k_wmma_gemm<<<dim3(gx, 512 / 64), TB, 0, stream>>>(h2b, bd1t, exp_b1, nullptr, s1e1, N, 128, 512, 1);
  k_wmma_gemm<<<dim3(gx, 128 / 64), TB, 0, stream>>>(s1e1, bd2t, exp_b2, nullptr, specb, N, 512, 128, 0);

  // ---- router: V path only (Q/K discarded in reference) --------------------
  k_wmma_gemm<<<dim3(gx, 1), TB, 0, stream>>>(specb, vwt, v_b, nullptr, veb, N, 128, 64, 0);
  k_wmma_gemm<<<dim3(gx, 1), TB, 0, stream>>>(veb, inwvt, in_b + 128, nullptr, vpb, N, 64, 64, 0);
  k_wmma_gemm<<<dim3(gx, 1), TB, 0, stream>>>(vpb, outwt, out_b, mha, nullptr, N, 64, 64, 0);

  k_layernorm64<<<cdiv(N, TB / 32), TB, 0, stream>>>(mha, ln_g, ln_b, attn, attnb, N);
  k_sel_softmax<<<cdiv(N, TB), TB, 0, stream>>>(attn, sel_w, sel_b, ew, N);

  // ---- router experts (block-diagonal) -------------------------------------
  k_wmma_gemm<<<dim3(gx, 512 / 64), TB, 0, stream>>>(attnb, rbd1t, rexp_b1, nullptr, s1e1, N, 64, 512, 1);
  k_wmma_gemm<<<dim3(gx, 512 / 64), TB, 0, stream>>>(s1e1, rbd2t, rexp_b2, eo, nullptr, N, 512, 512, 0);

  k_routed_combine<<<cdiv(N * 128, TB), TB, 0, stream>>>(eo, ew, x, (float*)d_out, N);
}
